// ProteinFeatures_42640435315064
// MI455X (gfx1250) — matrix operand order repaired and checked by reading, not compiled, not run on hardware
//
#include <hip/hip_runtime.h>
#include <hip/hip_bf16.h>
#include <float.h>

typedef __attribute__((ext_vector_type(16))) _Float16 v16h;
typedef __attribute__((ext_vector_type(8)))  _Float16 v8h;
typedef __attribute__((ext_vector_type(8)))  float    v8f;

#define BN 4
#define LN 2048
#define TOPK 48
#define FDIM 416      // 16 pos-enc + 25*16 RBF
#define EDIM 128
#define KSTEPS 13     // 416 / 32
#define FSTRIDE 424   // padded f16 row stride: 424*2 = 848 = 53*16 bytes (16B-aligned rows)
#define ESTRIDE 132   // padded f32 row stride for E tile

// ---------------------------------------------------------------------------
// Kernel 1: build 5-atom table (N,Ca,C,O,virtual Cb) and f16 copy of edge_w
// ---------------------------------------------------------------------------
__global__ void pf_prep_kernel(const float* __restrict__ X,
                               const float* __restrict__ edge_w,
                               float* __restrict__ atoms5,
                               _Float16* __restrict__ edgew_h) {
    int i = blockIdx.x * blockDim.x + threadIdx.x;
    const int nres = BN * LN;
    if (i < nres) {
        const float* xr = X + (size_t)i * 12;
        float N[3], Ca[3], C[3], O[3], b[3], c[3], a[3], Cb[3];
        #pragma unroll
        for (int q = 0; q < 3; ++q) { N[q]=xr[q]; Ca[q]=xr[3+q]; C[q]=xr[6+q]; O[q]=xr[9+q]; }
        #pragma unroll
        for (int q = 0; q < 3; ++q) { b[q] = Ca[q]-N[q]; c[q] = C[q]-Ca[q]; }
        a[0]=b[1]*c[2]-b[2]*c[1]; a[1]=b[2]*c[0]-b[0]*c[2]; a[2]=b[0]*c[1]-b[1]*c[0];
        #pragma unroll
        for (int q = 0; q < 3; ++q)
            Cb[q] = -0.58273431f*a[q] + 0.56802827f*b[q] - 0.54067466f*c[q] + Ca[q];
        float* o = atoms5 + (size_t)i * 15;
        #pragma unroll
        for (int q = 0; q < 3; ++q) { o[q]=N[q]; o[3+q]=Ca[q]; o[6+q]=C[q]; o[9+q]=O[q]; o[12+q]=Cb[q]; }
    }
    const int ne = EDIM * FDIM;
    const int nthr = gridDim.x * blockDim.x;
    for (int j = i; j < ne; j += nthr)
        edgew_h[j] = (_Float16)edge_w[j];
}

// ---------------------------------------------------------------------------
// Kernel 2: brute-force kNN on masked Ca-Ca distances, faithful to reference:
//   D = m2 * sqrt(d2 + 1e-6); D_adjust = D + (1-m2)*rowmax(D); top-48 smallest
//   (lowest-index tie break, matching lax.top_k stability)
// ---------------------------------------------------------------------------
__global__ void __launch_bounds__(256)
pf_knn_kernel(const float* __restrict__ X, const float* __restrict__ mask,
              int* __restrict__ Eidx, float* __restrict__ Dnb) {
    const int l = blockIdx.x, b = blockIdx.y, t = threadIdx.x;
    __shared__ float Drow[LN];
    __shared__ float rv[256];
    __shared__ int   ri[256];

    const float* Xb = X + (size_t)b * LN * 12;
    const float cax = Xb[l*12+3], cay = Xb[l*12+4], caz = Xb[l*12+5];
    const float mi = mask[b*LN + l];

    float lmax = -FLT_MAX;
    for (int j = t; j < LN; j += 256) {
        float dx = cax - Xb[j*12+3];
        float dy = cay - Xb[j*12+4];
        float dz = caz - Xb[j*12+5];
        float m2 = mi * mask[b*LN + j];
        float D  = m2 * sqrtf(dx*dx + dy*dy + dz*dz + 1e-6f);
        Drow[j] = D;
        lmax = fmaxf(lmax, D);
    }
    rv[t] = lmax; __syncthreads();
    for (int s = 128; s; s >>= 1) { if (t < s) rv[t] = fmaxf(rv[t], rv[t+s]); __syncthreads(); }
    const float rowmax = rv[0]; __syncthreads();

    for (int j = t; j < LN; j += 256) {
        float m2 = mi * mask[b*LN + j];
        Drow[j] += (1.0f - m2) * rowmax;
    }
    __syncthreads();

    for (int kk = 0; kk < TOPK; ++kk) {
        float bv = FLT_MAX; int bi = 0x7FFFFFFF;
        for (int j = t; j < LN; j += 256) {
            float v = Drow[j];
            if (v < bv || (v == bv && j < bi)) { bv = v; bi = j; }
        }
        rv[t] = bv; ri[t] = bi; __syncthreads();
        for (int s = 128; s; s >>= 1) {
            if (t < s) {
                float v2 = rv[t+s]; int i2 = ri[t+s];
                if (v2 < rv[t] || (v2 == rv[t] && i2 < ri[t])) { rv[t] = v2; ri[t] = i2; }
            }
            __syncthreads();
        }
        if (t == 0) {
            int sel = ri[0];
            Eidx[((size_t)b*LN + l)*TOPK + kk] = sel;
            Dnb [((size_t)b*LN + l)*TOPK + kk] = rv[0];
            Drow[sel] = FLT_MAX;
        }
        __syncthreads();
    }
}

// build v16h A/B fragment from two 16B-aligned 8-half chunks
__device__ __forceinline__ v16h pf_frag(const _Float16* lo_p, const _Float16* hi_p) {
    v8h lo = *(const v8h*)lo_p;
    v8h hi = *(const v8h*)hi_p;
    v16h r;
    #pragma unroll
    for (int i = 0; i < 8; ++i) { r[i] = lo[i]; r[8+i] = hi[i]; }
    return r;
}

// ---------------------------------------------------------------------------
// Kernel 3: one block per (b,l): 48-edge feature tile (pos-enc + 25x16 RBF)
// in LDS (f16), then 8 waves x 13 ksteps x 3 M-tiles of
// v_wmma_f32_16x16x32_f16 with the B fragment held in registers (3x reuse
// -> 1/3 the L2 traffic on edge_w), then fused LayerNorm (+shfl_xor
// butterflies) and store. E tile aliases the feature tile in LDS.
// ---------------------------------------------------------------------------
__global__ void __launch_bounds__(256)
pf_edge_kernel(const float* __restrict__ atoms5,
               const _Float16* __restrict__ edgew_h,
               const int* __restrict__ residue_idx,
               const int* __restrict__ chain_labels,
               const float* __restrict__ pe_w,
               const float* __restrict__ pe_b,
               const float* __restrict__ ln_scale,
               const float* __restrict__ ln_bias,
               const int* __restrict__ Eidx,
               const float* __restrict__ Dnb,
               float* __restrict__ Eout) {
    const int l = blockIdx.x;
    const int b = blockIdx.y;
    const int t = threadIdx.x;

    // fsh (48 x 424 f16 = 40704 B) and Etile (48 x 132 f32 = 25344 B) alias
    __shared__ __attribute__((aligned(16))) char smem[TOPK * FSTRIDE * 2];
    _Float16 (*fsh)[FSTRIDE] = (_Float16 (*)[FSTRIDE])smem;
    float    (*Etile)[ESTRIDE] = (float (*)[ESTRIDE])smem;

    __shared__ float selfA[16];
    __shared__ float nbA[TOPK][16];
    __shared__ int   nbj[TOPK];

    const size_t basebl = (size_t)b * LN + l;

    if (t < TOPK) nbj[t]   = Eidx[basebl*TOPK + t];
    if (t < 15)   selfA[t] = atoms5[basebl*15 + t];
    __syncthreads();
    for (int q = t; q < TOPK*15; q += 256) {
        int row = q / 15, c = q % 15;
        nbA[row][c] = atoms5[((size_t)b*LN + nbj[row])*15 + c];
    }
    __syncthreads();

    const int row16 = t >> 4;      // 0..15
    const int j16   = t & 15;      // RBF index / pos-enc channel
    const float mu      = 2.0f + (20.0f/15.0f) * (float)j16;
    const float inv_sig = 0.8f;    // 1/1.25
    const int ridx_l = residue_idx[b*LN + l];
    const int ch_l   = chain_labels[b*LN + l];

    // atom-pair tables: self atom pa vs neighbor atom pb; N=0,Ca=1,C=2,O=3,Cb=4
    const int pa[24] = {0,2,3,4,1,1,1,1,0,0,0,4,4,3,0,2,3,4,2,3,4,2,3,2};
    const int pb[24] = {0,2,3,4,0,2,3,4,2,3,4,2,3,2,1,1,1,1,0,0,0,4,4,3};

    #pragma unroll
    for (int rg = 0; rg < 3; ++rg) {
        const int row = rg*16 + row16;     // edge row 0..47
        const int jn  = nbj[row];
        // positional encoding: one-hot(66) @ pe_w == column lookup
        {
            int doff = ridx_l - residue_idx[b*LN + jn];
            int same = (ch_l == chain_labels[b*LN + jn]);
            int d = doff + 32; d = d < 0 ? 0 : (d > 64 ? 64 : d);
            int oh = same ? d : 65;
            fsh[row][j16] = (_Float16)(pe_w[j16*66 + oh] + pe_b[j16]);
        }
        // RBF slot 0: masked Ca-Ca neighbor distance from kNN
        {
            float Dv = Dnb[basebl*TOPK + row];
            float u = (Dv - mu) * inv_sig;
            fsh[row][16 + j16] = (_Float16)__expf(-u*u);
        }
        // 24 atom pairs
        #pragma unroll
        for (int p = 0; p < 24; ++p) {
            int ia = pa[p]*3, ib = pb[p]*3;
            float dx = selfA[ia+0] - nbA[row][ib+0];
            float dy = selfA[ia+1] - nbA[row][ib+1];
            float dz = selfA[ia+2] - nbA[row][ib+2];
            float Dv = sqrtf(dx*dx + dy*dy + dz*dz + 1e-6f);
            float u = (Dv - mu) * inv_sig;
            fsh[row][16 + (p+1)*16 + j16] = (_Float16)__expf(-u*u);
        }
    }
    __syncthreads();

    // ---- WMMA: wave w covers output cols [16w,16w+16); 3 M-tiles share B ----
    const int w    = t >> 5;
    const int lane = t & 31;
    const int ma   = lane & 15;                 // A row within tile (M)
    const int kb   = (lane & 16) ? 8 : 0;       // A K sub-offset per ISA layout
    const int ecol = w*16 + (lane & 15);        // B column (output channel)
    const _Float16* brow = edgew_h + (size_t)ecol * FDIM + ((lane & 16) ? 8 : 0);

    v8f acc0 = {}, acc1 = {}, acc2 = {};
    #pragma unroll
    for (int ks = 0; ks < KSTEPS; ++ks) {
        // B fragment: halfs[0..7]=K kb..kb+7, halfs[8..15]=K 16+kb..16+kb+7 of edge_w row (16B-aligned), 3x reuse
        v16h Bm = pf_frag(brow + ks*32, brow + ks*32 + 16);
        // A fragments: halfs[0..7]=K kb..kb+7, halfs[8..15]=K 16+kb..16+kb+7
        const _Float16* a0 = &fsh[ma     ][ks*32 + kb];
        const _Float16* a1 = &fsh[16 + ma][ks*32 + kb];
        const _Float16* a2 = &fsh[32 + ma][ks*32 + kb];
        v16h A0 = pf_frag(a0, a0 + 16);
        v16h A1 = pf_frag(a1, a1 + 16);
        v16h A2 = pf_frag(a2, a2 + 16);
        acc0 = __builtin_amdgcn_wmma_f32_16x16x32_f16(false, A0, false, Bm, (short)0, acc0, false, false);
        acc1 = __builtin_amdgcn_wmma_f32_16x16x32_f16(false, A1, false, Bm, (short)0, acc1, false, false);
        acc2 = __builtin_amdgcn_wmma_f32_16x16x32_f16(false, A2, false, Bm, (short)0, acc2, false, false);
    }
    __syncthreads();   // all waves done reading fsh; Etile aliases it

    // C/D layout: VGPR r -> row (r + 8*(lane>=16)), col (lane&15)
    {
        int mbase = (lane & 16) ? 8 : 0;
        int n = lane & 15;
        #pragma unroll
        for (int r = 0; r < 8; ++r) {
            Etile[     mbase + r][w*16 + n] = acc0[r];
            Etile[16 + mbase + r][w*16 + n] = acc1[r];
            Etile[32 + mbase + r][w*16 + n] = acc2[r];
        }
    }
    __syncthreads();

    // ---- LayerNorm over 128 channels per edge row (shfl_xor butterflies) ----
    #pragma unroll
    for (int rg = 0; rg < 3; ++rg) {
        const int row = rg*16 + row16;
        float s = 0.0f, s2 = 0.0f;
        #pragma unroll
        for (int i = 0; i < 8; ++i) {
            float v = Etile[row][j16 + 16*i];
            s += v; s2 += v*v;
        }
        #pragma unroll
        for (int m = 1; m < 16; m <<= 1) {
            s  += __shfl_xor(s,  m, 32);
            s2 += __shfl_xor(s2, m, 32);
        }
        float mean = s  * (1.0f/128.0f);
        float var  = s2 * (1.0f/128.0f) - mean*mean;
        float rstd = rsqrtf(var + 1e-5f);
        float* orow = Eout + (basebl*TOPK + row) * EDIM;
        #pragma unroll
        for (int i = 0; i < 8; ++i) {
            int e = j16 + 16*i;
            orow[e] = (Etile[row][e] - mean) * rstd * ln_scale[e] + ln_bias[e];
        }
    }
}

// ---------------------------------------------------------------------------
extern "C" void kernel_launch(void* const* d_in, const int* in_sizes, int n_in,
                              void* d_out, int out_size, void* d_ws, size_t ws_size,
                              hipStream_t stream) {
    const float* X            = (const float*)d_in[0];
    const float* mask         = (const float*)d_in[1];
    const int*   residue_idx  = (const int*)d_in[2];
    const int*   chain_labels = (const int*)d_in[3];
    const float* pe_w         = (const float*)d_in[4];
    const float* pe_b         = (const float*)d_in[5];
    const float* edge_w       = (const float*)d_in[6];
    const float* ln_scale     = (const float*)d_in[7];
    const float* ln_bias      = (const float*)d_in[8];

    float* Eout  = (float*)d_out;
    int*   EidxO = (int*)((float*)d_out + (size_t)BN * LN * TOPK * EDIM);

    // workspace layout (all offsets 16B-aligned)
    char*     ws     = (char*)d_ws;
    float*    atoms5 = (float*)ws;                                   // BN*LN*15 f32
    float*    Dnb    = (float*)(ws + (size_t)BN*LN*15*4);            // BN*LN*48 f32
    _Float16* edgewh = (_Float16*)(ws + (size_t)BN*LN*15*4
                                      + (size_t)BN*LN*TOPK*4);       // 128*416 f16

    pf_prep_kernel<<<32, 256, 0, stream>>>(X, edge_w, atoms5, edgewh);
    pf_knn_kernel<<<dim3(LN, BN), 256, 0, stream>>>(X, mask, EidxO, Dnb);
    pf_edge_kernel<<<dim3(LN, BN), 256, 0, stream>>>(
        atoms5, edgewh, residue_idx, chain_labels, pe_w, pe_b,
        ln_scale, ln_bias, EidxO, Dnb, Eout);
}
